// ClassConditionedSpatialGatedFusionClassifier_90022514524602
// MI455X (gfx1250) — compile-verified, matrix operand order checked by reading, not run on hardware
//
#include <hip/hip_runtime.h>
#include <hip/hip_bf16.h>
#include <math.h>

#define BB   8192
#define DVV  1024
#define DTT  768
#define DXC  1797
#define HHH  256
#define NCLS 5
#define GHID 128
#define KNN  8
#define FDIM 512
#define GIN  773
#define GPAD 800
#define LNEPS 1e-5f
#define EWC  0.01f

typedef __attribute__((ext_vector_type(16))) _Float16 v16h;
typedef __attribute__((ext_vector_type(8)))  _Float16 v8h;
typedef __attribute__((ext_vector_type(8)))  float    v8f;

typedef __attribute__((ext_vector_type(4))) unsigned int u32x4;
typedef __attribute__((ext_vector_type(8))) int          i32x8;
typedef __attribute__((ext_vector_type(4))) int          i32x4;

#if __has_builtin(__builtin_amdgcn_tensor_load_to_lds) && __has_builtin(__builtin_amdgcn_s_wait_tensorcnt)
#define HAVE_TDM 1
#else
#define HAVE_TDM 0
#endif

// ---------------------------------------------------------------------------
// Generic WMMA GEMM:  C(M x N) = epilogue( A(M x K) @ W(K x N) )
// A, W, C in f16; accumulate f32. Block tile 128x64, 8 waves, wave tile 32x32.
// A tile is staged into LDS by the Tensor Data Mover (wave 0 issues the DMA,
// TENSORcnt-waited, workgroup barrier publishes) with pad_enable producing the
// 40-half row stride; B tile is staged transposed by all threads meanwhile.
// Epilogue: v = acc + bias[n]; v = v*scale[n]*smul + shift[n]; relu; resid+alpha*v
// Requires: M % 128 == 0, N % 64 == 0, K % 32 == 0 (true for all call sites).
// ---------------------------------------------------------------------------
__launch_bounds__(256)
__global__ void wmma_gemm_kernel(
    const _Float16* __restrict__ A, int lda,
    const _Float16* __restrict__ W, int ldw,
    const float* __restrict__ bias,
    const float* __restrict__ scale, const float* __restrict__ shift, float smul,
    const _Float16* __restrict__ resid, float alpha,
    _Float16* __restrict__ Cm, int ldc,
    int N, int Ktot, int relu)
{
    __shared__ _Float16 sA[128 * 40];   // 128 rows x 32 K (stride 40 halves = 80B)
    __shared__ _Float16 sBt[64 * 40];   // 64 cols  x 32 K, transposed

    const int t    = threadIdx.x;
    const int lane = t & 31;
    const int wv   = t >> 5;
    const int wr   = wv & 3;           // wave row 0..3 (32 rows each)
    const int wc   = wv >> 2;          // wave col 0..1 (32 cols each)
    const int nblk = N >> 6;
    const int bm   = (int)blockIdx.x / nblk;
    const int bn   = (int)blockIdx.x % nblk;
    const int m0   = bm << 7;
    const int n0   = bn << 6;

    v8f acc00 = {}, acc01 = {}, acc10 = {}, acc11 = {};

    const int koff = (lane < 16) ? 0 : 8;

#if HAVE_TDM
    // Loop-invariant pieces of the A-tile tensor descriptor (D#).
    const unsigned ldsA = (unsigned)(unsigned long long)(void*)&sA[0];
    const unsigned long long gabase =
        (unsigned long long)(const void*)(A + (size_t)m0 * lda);
#endif

    for (int k0 = 0; k0 < Ktot; k0 += 32) {
#if HAVE_TDM
        // --- stage A tile (128 rows x 32 halves) via Tensor Data Mover ---
        if (wv == 0) {
            const unsigned long long ga = gabase + (unsigned long long)k0 * 2ull;
            u32x4 g0;
            g0[0] = 1u;                                    // count=1, user mode
            g0[1] = ldsA;                                  // LDS dest (bytes)
            g0[2] = (unsigned)ga;                          // global_addr[31:0]
            g0[3] = ((unsigned)(ga >> 32) & 0x01FFFFFFu)   // global_addr[56:32]
                    | 0x80000000u;                         // type=2 ("image")
            i32x8 g1;
            g1[0] = (int)((1u << 16)        // data_size = 2 bytes
                        | (1u << 20)        // pad_enable
                        | (3u << 22)        // pad_interval: 16 DWORDs (=64B row)
                        | (3u << 25));      // pad_amount:   4 DWORDs (=16B pad)
            g1[1] = (int)(((unsigned)lda & 0xFFFFu) << 16);          // tensor_dim0 lo
            g1[2] = (int)((((unsigned)lda >> 16) & 0xFFFFu)
                        | (((unsigned)BB & 0xFFFFu) << 16));         // dim0 hi | dim1 lo
            g1[3] = (int)((((unsigned)BB >> 16) & 0xFFFFu)
                        | (32u << 16));                              // dim1 hi | tile_dim0=32
            g1[4] = 128;                                             // tile_dim1=128
            g1[5] = lda;                                             // dim0_stride lo32
            g1[6] = 0;
            g1[7] = 0;
            const i32x4 gz4 = {0, 0, 0, 0};                          // 2D: groups 2/3 zero
            const i32x8 gz8 = {0, 0, 0, 0, 0, 0, 0, 0};
            __builtin_amdgcn_tensor_load_to_lds(g0, g1, gz4, gz4, gz8, 0);
        }
#else
        // --- stage A tile (128x32 halves) via 16B vector loads ---
        #pragma unroll
        for (int j = 0; j < 2; ++j) {
            int c  = t * 2 + j;             // 0..511 chunks of 8 halves
            int ar = c >> 2;
            int ac = (c & 3) << 3;
            v8h av = *(const v8h*)(A + (size_t)(m0 + ar) * lda + k0 + ac);
            *(v8h*)(sA + ar * 40 + ac) = av;
        }
#endif
        // --- stage B tile (32x64) transposed into sBt[n][k] (overlaps TDM) ---
        {
            int br = t >> 3;
            int bc = (t & 7) << 3;
            v8h bv = *(const v8h*)(W + (size_t)(k0 + br) * ldw + n0 + bc);
            #pragma unroll
            for (int j = 0; j < 8; ++j) sBt[(bc + j) * 40 + br] = bv[j];
        }
        if (k0 + 32 < Ktot)
            __builtin_prefetch(W + (size_t)(k0 + 32 + (t >> 3)) * ldw + n0, 0, 1);
#if HAVE_TDM
        if (wv == 0) __builtin_amdgcn_s_wait_tensorcnt(0);
#endif
        __syncthreads();

        auto ldA = [&](int mb) -> v16h {
            const _Float16* p = sA + (mb + (lane & 15)) * 40;
            v8h lo = *(const v8h*)(p + koff);
            v8h hi = *(const v8h*)(p + 16 + koff);
            v16h f;
            #pragma unroll
            for (int q = 0; q < 8; ++q) { f[q] = lo[q]; f[q + 8] = hi[q]; }
            return f;
        };
        auto ldB = [&](int nb) -> v16h {
            const _Float16* p = sBt + (nb + (lane & 15)) * 40;
            v8h lo = *(const v8h*)(p + koff);
            v8h hi = *(const v8h*)(p + 16 + koff);
            v16h f;
            #pragma unroll
            for (int q = 0; q < 8; ++q) { f[q] = lo[q]; f[q + 8] = hi[q]; }
            return f;
        };

        v16h a0 = ldA((wr << 5) + 0);
        v16h a1 = ldA((wr << 5) + 16);
        v16h b0 = ldB((wc << 5) + 0);
        v16h b1 = ldB((wc << 5) + 16);

        acc00 = __builtin_amdgcn_wmma_f32_16x16x32_f16(false, a0, false, b0, (short)0, acc00, false, false);
        acc01 = __builtin_amdgcn_wmma_f32_16x16x32_f16(false, a0, false, b1, (short)0, acc01, false, false);
        acc10 = __builtin_amdgcn_wmma_f32_16x16x32_f16(false, a1, false, b0, (short)0, acc10, false, false);
        acc11 = __builtin_amdgcn_wmma_f32_16x16x32_f16(false, a1, false, b1, (short)0, acc11, false, false);

        __syncthreads();
    }

    const int cb   = lane & 15;
    const int rsel = (lane < 16) ? 0 : 8;

    auto storeFrag = [&](v8f acc, int mi, int nj) {
        int gcol = n0 + (wc << 5) + nj + cb;
        float bv = bias ? bias[gcol] : 0.0f;
        float sc = scale ? scale[gcol] * smul : 1.0f;
        float sh = shift ? shift[gcol] : 0.0f;
        #pragma unroll
        for (int r = 0; r < 8; ++r) {
            int grow = m0 + (wr << 5) + mi + r + rsel;
            float v = acc[r] + bv;
            v = v * sc + sh;
            if (relu) v = fmaxf(v, 0.0f);
            if (resid) v = (float)resid[(size_t)grow * ldc + gcol] + alpha * v;
            Cm[(size_t)grow * ldc + gcol] = (_Float16)v;
        }
    };
    storeFrag(acc00, 0, 0);
    storeFrag(acc01, 0, 16);
    storeFrag(acc10, 16, 0);
    storeFrag(acc11, 16, 16);
}

// ---------------------------------------------------------------------------
// LayerNorm prep: per row, LN the vision (1024) and text (768) slices with
// gamma/beta, store f16 into hx[B x 1792]; also extract coords + image id.
// ---------------------------------------------------------------------------
__launch_bounds__(256)
__global__ void ln_prep_kernel(const float* __restrict__ x,
                               const float* __restrict__ g_v, const float* __restrict__ b_v,
                               const float* __restrict__ g_t, const float* __restrict__ b_t,
                               _Float16* __restrict__ hx,
                               float* __restrict__ coords, int* __restrict__ imgi)
{
    __shared__ float s1[256], s2[256];
    const int i = blockIdx.x, t = threadIdx.x;
    const float* xr = x + (size_t)i * DXC;

    // vision slice
    float su = 0.f, sq = 0.f;
    for (int c = t; c < DVV; c += 256) { float v = xr[c]; su += v; sq += v * v; }
    s1[t] = su; s2[t] = sq; __syncthreads();
    for (int s = 128; s > 0; s >>= 1) { if (t < s) { s1[t] += s1[t + s]; s2[t] += s2[t + s]; } __syncthreads(); }
    float mean = s1[0] / (float)DVV;
    float var  = s2[0] / (float)DVV - mean * mean;
    float rs   = rsqrtf(var + LNEPS);
    for (int c = t; c < DVV; c += 256)
        hx[(size_t)i * 1792 + c] = (_Float16)(((xr[c] - mean) * rs) * g_v[c] + b_v[c]);
    __syncthreads();

    // text slice
    su = 0.f; sq = 0.f;
    for (int c = t; c < DTT; c += 256) { float v = xr[DVV + c]; su += v; sq += v * v; }
    s1[t] = su; s2[t] = sq; __syncthreads();
    for (int s = 128; s > 0; s >>= 1) { if (t < s) { s1[t] += s1[t + s]; s2[t] += s2[t + s]; } __syncthreads(); }
    mean = s1[0] / (float)DTT;
    var  = s2[0] / (float)DTT - mean * mean;
    rs   = rsqrtf(var + LNEPS);
    for (int c = t; c < DTT; c += 256)
        hx[(size_t)i * 1792 + DVV + c] = (_Float16)(((xr[DVV + c] - mean) * rs) * g_t[c] + b_t[c]);

    if (t == 0) {
        coords[i * 2 + 0] = xr[DVV + DTT + 0];
        coords[i * 2 + 1] = xr[DVV + DTT + 1];
        imgi[i] = (int)xr[DVV + DTT + 4];
    }
}

// ---------------------------------------------------------------------------
// KNN top-8 with same-image masking + softmax weights (shared by both passes)
// ---------------------------------------------------------------------------
__device__ __forceinline__ void ins8(float v, int id, float* vs, int* is) {
    if (v <= vs[7]) return;
    int p = 7;
    while (p > 0 && vs[p - 1] < v) { vs[p] = vs[p - 1]; is[p] = is[p - 1]; --p; }
    vs[p] = v; is[p] = id;
}

__launch_bounds__(256)
__global__ void knn_topk_kernel(const float* __restrict__ coords, const int* __restrict__ imgi,
                                int* __restrict__ knn_idx, float* __restrict__ knn_w)
{
    __shared__ float shv[256 * 8];
    __shared__ int   shi[256 * 8];
    __shared__ int   scnt[256];
    const int i = blockIdx.x, t = threadIdx.x;
    const float cx = coords[i * 2], cy = coords[i * 2 + 1];
    const int   im = imgi[i];

    float sv[8], av[8]; int si[8], ai[8];
    #pragma unroll
    for (int k = 0; k < 8; ++k) { sv[k] = -3.0e38f; av[k] = -3.0e38f; si[k] = -1; ai[k] = -1; }
    int cnt = 0;
    for (int j = t; j < BB; j += 256) {
        if (j == i) continue;
        float dx = coords[j * 2] - cx;
        float dy = coords[j * 2 + 1] - cy;
        float sim = -sqrtf(fmaxf(dx * dx + dy * dy, 0.0f));
        ins8(sim, j, av, ai);
        if (imgi[j] == im) { ++cnt; ins8(sim, j, sv, si); }
    }
    scnt[t] = cnt; __syncthreads();
    for (int s = 128; s > 0; s >>= 1) { if (t < s) scnt[t] += scnt[t + s]; __syncthreads(); }
    const bool useSame = scnt[0] > 0;

    #pragma unroll
    for (int k = 0; k < 8; ++k) {
        shv[t * 8 + k] = useSame ? sv[k] : av[k];
        shi[t * 8 + k] = useSame ? si[k] : ai[k];
    }
    __syncthreads();

    // pairwise merge of sorted-descending 8-lists
    for (int s = 128; s > 0; s >>= 1) {
        if (t < s) {
            float a[8], b[8]; int ia[8], ib[8];
            #pragma unroll
            for (int k = 0; k < 8; ++k) {
                a[k] = shv[t * 8 + k];        ia[k] = shi[t * 8 + k];
                b[k] = shv[(t + s) * 8 + k];  ib[k] = shi[(t + s) * 8 + k];
            }
            float o[8]; int io[8]; int p = 0, q = 0;
            #pragma unroll
            for (int k = 0; k < 8; ++k) {
                bool ta = (q >= 8) || (p < 8 && a[p] >= b[q]);
                if (ta) { o[k] = a[p]; io[k] = ia[p]; ++p; }
                else    { o[k] = b[q]; io[k] = ib[q]; ++q; }
            }
            #pragma unroll
            for (int k = 0; k < 8; ++k) { shv[t * 8 + k] = o[k]; shi[t * 8 + k] = io[k]; }
        }
        __syncthreads();
    }

    if (t == 0) {
        float m = shv[0];
        float e[8], ssum = 0.f;
        #pragma unroll
        for (int k = 0; k < 8; ++k) { e[k] = __expf(shv[k] - m); ssum += e[k]; }
        #pragma unroll
        for (int k = 0; k < 8; ++k) {
            knn_w[i * 8 + k]   = e[k] / ssum;
            knn_idx[i * 8 + k] = shi[k] < 0 ? 0 : shi[k];
        }
    }
}

__launch_bounds__(256)
__global__ void knn_gather_kernel(const _Float16* __restrict__ feat,
                                  const int* __restrict__ knn_idx, const float* __restrict__ knn_w,
                                  _Float16* __restrict__ out)
{
    __shared__ int   sid[8];
    __shared__ float sw[8];
    const int i = blockIdx.x, t = threadIdx.x;
    if (t < 8) { sid[t] = knn_idx[i * 8 + t]; sw[t] = knn_w[i * 8 + t]; }
    __syncthreads();
    for (int c = t; c < FDIM; c += 256) {
        float acc = 0.f;
        #pragma unroll
        for (int k = 0; k < 8; ++k) acc += sw[k] * (float)feat[(size_t)sid[k] * FDIM + c];
        out[(size_t)i * FDIM + c] = (_Float16)acc;
    }
}

// ---------------------------------------------------------------------------
// Wave-per-row small projections
// ---------------------------------------------------------------------------
__launch_bounds__(256)
__global__ void class_prior_kernel(const _Float16* __restrict__ base,
                                   const float* __restrict__ Wcp, const float* __restrict__ bcp,
                                   _Float16* __restrict__ gate_in)
{
    const int lane = threadIdx.x & 31, wv = threadIdx.x >> 5;
    const int i = blockIdx.x * 8 + wv;
    float acc[NCLS] = {};
    const _Float16* ar = base + (size_t)i * FDIM;
    for (int k = lane; k < FDIM; k += 32) {
        float a = (float)ar[k];
        #pragma unroll
        for (int n = 0; n < NCLS; ++n) acc[n] += a * Wcp[k * NCLS + n];
    }
    #pragma unroll
    for (int n = 0; n < NCLS; ++n)
        for (int off = 16; off > 0; off >>= 1) acc[n] += __shfl_xor(acc[n], off);
    if (lane == 0) {
        float m = -3.0e38f;
        #pragma unroll
        for (int n = 0; n < NCLS; ++n) { acc[n] += bcp[n]; m = fmaxf(m, acc[n]); }
        float e[NCLS], s = 0.f;
        #pragma unroll
        for (int n = 0; n < NCLS; ++n) { e[n] = __expf(acc[n] - m); s += e[n]; }
        #pragma unroll
        for (int n = 0; n < NCLS; ++n)
            gate_in[(size_t)i * GPAD + FDIM + n] = (_Float16)(e[n] / s);
    }
}

__launch_bounds__(256)
__global__ void pack_gate_kernel(const _Float16* __restrict__ base,
                                 const _Float16* __restrict__ spatial,
                                 _Float16* __restrict__ gate_in)
{
    const int i = blockIdx.x, t = threadIdx.x;
    for (int c = t; c < GPAD; c += 256) {
        if (c >= FDIM && c < FDIM + NCLS) continue;   // class prior written elsewhere
        _Float16 v;
        if (c < FDIM)       v = base[(size_t)i * FDIM + c];
        else if (c < GIN)   v = spatial[(size_t)i * HHH + (c - (FDIM + NCLS))];
        else                v = (_Float16)0.0f;
        gate_in[(size_t)i * GPAD + c] = v;
    }
}

__launch_bounds__(256)
__global__ void gate_kernel(const _Float16* __restrict__ gh,
                            const float* __restrict__ Wg2, const float* __restrict__ bg2,
                            float* __restrict__ gprobs, float* __restrict__ ent_out)
{
    const int lane = threadIdx.x & 31, wv = threadIdx.x >> 5;
    const int i = blockIdx.x * 8 + wv;
    float a0 = 0.f, a1 = 0.f;
    const _Float16* ar = gh + (size_t)i * GHID;
    for (int k = lane; k < GHID; k += 32) {
        float a = (float)ar[k];
        a0 += a * Wg2[k * 2];
        a1 += a * Wg2[k * 2 + 1];
    }
    for (int off = 16; off > 0; off >>= 1) { a0 += __shfl_xor(a0, off); a1 += __shfl_xor(a1, off); }
    if (lane == 0) {
        a0 += bg2[0]; a1 += bg2[1];
        float m  = fmaxf(a0, a1);
        float e0 = __expf(a0 - m), e1 = __expf(a1 - m);
        float s  = e0 + e1;
        float p0 = e0 / s, p1 = e1 / s;
        gprobs[i * 2 + 0] = p0;
        gprobs[i * 2 + 1] = p1;
        float ent = -(p0 * __logf(p0 + 1e-8f) + p1 * __logf(p1 + 1e-8f));
        atomicAdd(ent_out, ent * (EWC / (float)BB));
    }
}

__launch_bounds__(256)
__global__ void logits_kernel(const _Float16* __restrict__ h,
                              const float* __restrict__ Wc2, const float* __restrict__ bc2,
                              float* __restrict__ out)
{
    const int lane = threadIdx.x & 31, wv = threadIdx.x >> 5;
    const int i = blockIdx.x * 8 + wv;
    float acc[NCLS] = {};
    const _Float16* ar = h + (size_t)i * HHH;
    for (int k = lane; k < HHH; k += 32) {
        float a = (float)ar[k];
        #pragma unroll
        for (int n = 0; n < NCLS; ++n) acc[n] += a * Wc2[k * NCLS + n];
    }
    #pragma unroll
    for (int n = 0; n < NCLS; ++n)
        for (int off = 16; off > 0; off >>= 1) acc[n] += __shfl_xor(acc[n], off);
    if (lane == 0) {
        #pragma unroll
        for (int n = 0; n < NCLS; ++n) out[(size_t)i * NCLS + n] = acc[n] + bc2[n];
    }
}

// ---------------------------------------------------------------------------
// Elementwise helpers
// ---------------------------------------------------------------------------
__global__ void fuse_gate_kernel(const _Float16* __restrict__ base,
                                 const float* __restrict__ gp,
                                 _Float16* __restrict__ fused)
{
    const int t = blockIdx.x * 256 + threadIdx.x;    // over B*512
    const int i = t >> 9, c = t & 511;
    float g = gp[i * 2 + (c >= 256 ? 1 : 0)];
    fused[t] = (_Float16)((float)base[t] * g);
}

__global__ void f32_to_f16_kernel(const float* __restrict__ s, _Float16* __restrict__ d, int n)
{
    const int t = blockIdx.x * 256 + threadIdx.x;
    if (t < n) d[t] = (_Float16)s[t];
}

__global__ void convert_pad_g1_kernel(const float* __restrict__ s, _Float16* __restrict__ d)
{
    const int t = blockIdx.x * 256 + threadIdx.x;    // over GPAD*GHID
    if (t >= GPAD * GHID) return;
    const int k = t / GHID, n = t % GHID;
    d[t] = (_Float16)(k < GIN ? s[k * GHID + n] : 0.0f);
}

// ---------------------------------------------------------------------------
extern "C" void kernel_launch(void* const* d_in, const int* in_sizes, int n_in,
                              void* d_out, int out_size, void* d_ws, size_t ws_size,
                              hipStream_t stream) {
    (void)in_sizes; (void)n_in; (void)out_size; (void)ws_size;

    const float* x      = (const float*)d_in[0];
    const float* ln_v_g = (const float*)d_in[1];
    const float* ln_v_b = (const float*)d_in[2];
    const float* ln_t_g = (const float*)d_in[3];
    const float* ln_t_b = (const float*)d_in[4];
    const float* W_v    = (const float*)d_in[5];
    const float* b_v    = (const float*)d_in[6];
    const float* W_t    = (const float*)d_in[7];
    const float* b_t    = (const float*)d_in[8];
    const float* W_cp   = (const float*)d_in[9];
    const float* b_cp   = (const float*)d_in[10];
    const float* W_ctx1 = (const float*)d_in[11];
    const float* b_ctx1 = (const float*)d_in[12];
    const float* W_ctx2 = (const float*)d_in[13];
    const float* b_ctx2 = (const float*)d_in[14];
    const float* W_g1   = (const float*)d_in[15];
    const float* b_g1   = (const float*)d_in[16];
    const float* W_g2   = (const float*)d_in[17];
    const float* b_g2   = (const float*)d_in[18];
    const float* W_gu1  = (const float*)d_in[19];
    const float* b_gu1  = (const float*)d_in[20];
    const float* W_gu2  = (const float*)d_in[21];
    const float* b_gu2  = (const float*)d_in[22];
    const float* W_c1   = (const float*)d_in[23];
    const float* b_c1   = (const float*)d_in[24];
    const float* bn_g   = (const float*)d_in[25];
    const float* bn_b   = (const float*)d_in[26];
    const float* W_c2   = (const float*)d_in[27];
    const float* b_c2   = (const float*)d_in[28];

    float* out_logits = (float*)d_out;
    float* out_ent    = out_logits + (size_t)BB * NCLS;

    // ---- carve workspace ----
    uint8_t* wsp = (uint8_t*)d_ws;
    size_t off = 0;
    auto take = [&](size_t bytes) -> void* {
        void* p = wsp + off;
        off += (bytes + 255) & ~(size_t)255;
        return p;
    };
    _Float16* hx      = (_Float16*)take((size_t)BB * 1792 * 2);
    _Float16* base    = (_Float16*)take((size_t)BB * FDIM * 2);
    _Float16* neigh   = (_Float16*)take((size_t)BB * FDIM * 2);  // reused for spatial_update
    _Float16* tmp256  = (_Float16*)take((size_t)BB * HHH * 2);   // ctx hidden, then h
    _Float16* spatial = (_Float16*)take((size_t)BB * HHH * 2);
    _Float16* gate_in = (_Float16*)take((size_t)BB * GPAD * 2);  // reused as u1 (B x 512)
    _Float16* g_h     = (_Float16*)take((size_t)BB * GHID * 2);
    _Float16* fused   = (_Float16*)take((size_t)BB * FDIM * 2);
    float*    coords  = (float*)take((size_t)BB * 2 * 4);
    int*      imgi    = (int*)take((size_t)BB * 4);
    int*      knn_idx = (int*)take((size_t)BB * KNN * 4);
    float*    knn_w   = (float*)take((size_t)BB * KNN * 4);
    float*    gprobs  = (float*)take((size_t)BB * 2 * 4);
    _Float16* Wvh     = (_Float16*)take((size_t)DVV * HHH * 2);
    _Float16* Wth     = (_Float16*)take((size_t)DTT * HHH * 2);
    _Float16* Wctx1h  = (_Float16*)take((size_t)FDIM * HHH * 2);
    _Float16* Wctx2h  = (_Float16*)take((size_t)HHH * HHH * 2);
    _Float16* Wg1p    = (_Float16*)take((size_t)GPAD * GHID * 2);
    _Float16* Wgu1h   = (_Float16*)take((size_t)FDIM * FDIM * 2);
    _Float16* Wgu2h   = (_Float16*)take((size_t)FDIM * FDIM * 2);
    _Float16* Wc1h    = (_Float16*)take((size_t)FDIM * HHH * 2);

    const float bninv = 1.0f / sqrtf(1.0f + 1e-5f);

    // entropy accumulator reset
    (void)hipMemsetAsync(out_ent, 0, sizeof(float), stream);

    // weight conversions f32 -> f16
    auto cvt = [&](const float* s, _Float16* d, int n) {
        f32_to_f16_kernel<<<(n + 255) / 256, 256, 0, stream>>>(s, d, n);
    };
    cvt(W_v,    Wvh,    DVV * HHH);
    cvt(W_t,    Wth,    DTT * HHH);
    cvt(W_ctx1, Wctx1h, FDIM * HHH);
    cvt(W_ctx2, Wctx2h, HHH * HHH);
    cvt(W_gu1,  Wgu1h,  FDIM * FDIM);
    cvt(W_gu2,  Wgu2h,  FDIM * FDIM);
    cvt(W_c1,   Wc1h,   FDIM * HHH);
    convert_pad_g1_kernel<<<(GPAD * GHID + 255) / 256, 256, 0, stream>>>(W_g1, Wg1p);

    // LN + feature extraction
    ln_prep_kernel<<<BB, 256, 0, stream>>>(x, ln_v_g, ln_v_b, ln_t_g, ln_t_b, hx, coords, imgi);

    auto gemm = [&](const _Float16* A, int lda, const _Float16* W, int ldw,
                    const float* bias, const float* scale, const float* shift, float smul,
                    const _Float16* resid, float alpha,
                    _Float16* C, int ldc, int N, int K, int relu) {
        dim3 grid((BB / 128) * (N / 64));
        wmma_gemm_kernel<<<grid, 256, 0, stream>>>(A, lda, W, ldw, bias, scale, shift, smul,
                                                   resid, alpha, C, ldc, N, K, relu);
    };

    // base_fused = [ LN(xv)@W_v + b_v , LN(xt)@W_t + b_t ]
    gemm(hx,        1792, Wvh, HHH, b_v, nullptr, nullptr, 1.f, nullptr, 0.f, base,       FDIM, HHH, DVV, 0);
    gemm(hx + DVV,  1792, Wth, HHH, b_t, nullptr, nullptr, 1.f, nullptr, 0.f, base + HHH, FDIM, HHH, DTT, 0);

    // KNN top-8 (shared by both aggregation passes)
    knn_topk_kernel<<<BB, 256, 0, stream>>>(coords, imgi, knn_idx, knn_w);
    knn_gather_kernel<<<BB, 256, 0, stream>>>(base, knn_idx, knn_w, neigh);

    // spatial_token = relu(neigh @ W_ctx1 + b) @ W_ctx2 + b
    gemm(neigh,  FDIM, Wctx1h, HHH, b_ctx1, nullptr, nullptr, 1.f, nullptr, 0.f, tmp256,  HHH, HHH, FDIM, 1);
    gemm(tmp256, HHH,  Wctx2h, HHH, b_ctx2, nullptr, nullptr, 1.f, nullptr, 0.f, spatial, HHH, HHH, HHH,  0);

    // gate input assembly (base | softmax(base@W_cp+b) | spatial | zero-pad)
    class_prior_kernel<<<BB / 8, 256, 0, stream>>>(base, W_cp, b_cp, gate_in);
    pack_gate_kernel<<<BB, 256, 0, stream>>>(base, spatial, gate_in);

    // gate MLP + softmax + entropy
    gemm(gate_in, GPAD, Wg1p, GHID, b_g1, nullptr, nullptr, 1.f, nullptr, 0.f, g_h, GHID, GHID, GPAD, 1);
    gate_kernel<<<BB / 8, 256, 0, stream>>>(g_h, W_g2, b_g2, gprobs, out_ent);

    // fused = [fv*g0, ft*g1], second KNN aggregation (reuse indices/weights)
    fuse_gate_kernel<<<(BB * FDIM) / 256, 256, 0, stream>>>(base, gprobs, fused);
    knn_gather_kernel<<<BB, 256, 0, stream>>>(fused, knn_idx, knn_w, neigh);

    // fused += 0.5 * ( relu(supd@W_gu1+b) @ W_gu2 + b )   (u1 lives in gate_in buffer)
    _Float16* u1 = gate_in;
    gemm(neigh, FDIM, Wgu1h, FDIM, b_gu1, nullptr, nullptr, 1.f, nullptr, 0.f, u1,    FDIM, FDIM, FDIM, 1);
    gemm(u1,    FDIM, Wgu2h, FDIM, b_gu2, nullptr, nullptr, 1.f, fused,   0.5f, fused, FDIM, FDIM, FDIM, 0);

    // h = relu( (fused@W_c1 + b_c1) * bn_g/sqrt(1+eps) + bn_b )
    gemm(fused, FDIM, Wc1h, HHH, b_c1, bn_g, bn_b, bninv, nullptr, 0.f, tmp256, HHH, HHH, FDIM, 1);

    // logits = h @ W_c2 + b_c2
    logits_kernel<<<BB / 8, 256, 0, stream>>>(tmp256, W_c2, b_c2, out_logits);
}
